// UGRNNNet_69939247448791
// MI455X (gfx1250) — matrix operand order repaired and checked by reading, not compiled, not run on hardware
//
#include <hip/hip_runtime.h>

typedef _Float16 half_t;
typedef __attribute__((ext_vector_type(8)))  _Float16 v8h;
typedef __attribute__((ext_vector_type(16))) _Float16 v16h;
typedef __attribute__((ext_vector_type(8)))  float    v8f;
typedef __attribute__((ext_vector_type(4)))  float    v4f;

#define BB   64
#define SS   512
#define FIN  256
#define HH   512
#define FOUT 128

// padded LDS row strides (halves): row stride in dwords % 64 == 4 -> conflict-free A loads
constexpr int XS = FIN + 8;  // 264
constexpr int HS = HH + 8;   // 520

__device__ __forceinline__ v16h cat16(v8h lo, v8h hi) {
  return __builtin_shufflevector(lo, hi, 0,1,2,3,4,5,6,7,8,9,10,11,12,13,14,15);
}

__device__ __forceinline__ float sigmoidf_(float x) {
  return 1.0f / (1.0f + __expf(-x));
}

// Accumulate over one contiguous A-region.
//   wrow  : weight pointer already at  WT + ncol*K + laneKoff + region_k_start
//   pa0   : A-region base already at   abase + (lane&15)*astride + laneC0
// A-tile layout (16x32 f16): lanes 0-15 row M=lane, K chunks [0:8)+[16:24);
//                            lanes 16-31 row M=lane-16, chunks [8:16)+[24:32).
// NOTE: unroll 1 is deliberate — full unrolling makes the compiler batch-hoist
// all weight global loads and spill the B-tiles through scratch.
__device__ __forceinline__ void gemm_region(const half_t* __restrict__ wrow,
                                            const half_t* pa0, int astride,
                                            int nkk, v8f acc[4]) {
  #pragma unroll 1
  for (int kk = 0; kk < nkk; ++kk) {
    const int k0 = kk * 32;
    v16h b = cat16(*(const v8h*)(wrow + k0), *(const v8h*)(wrow + k0 + 16));
    __builtin_prefetch(wrow + k0 + 32, 0, 1);  // global_prefetch_b8 (streams next chunk)
    #pragma unroll
    for (int m = 0; m < 4; ++m) {
      const half_t* p = pa0 + (m * 16) * astride + k0;
      v16h a = cat16(*(const v8h*)p, *(const v8h*)(p + 16));
      acc[m] = __builtin_amdgcn_wmma_f32_16x16x32_f16(false, a, false, b, (short)0, acc[m], false, false);
    }
  }
}

// One recurrent layer step: h = sigmoid(comb@Wz+bz)*h + (1-sigmoid)*tanh(comb@Wh+bh)
// comb = [region0 (len0 cols, stride s0) | lh (stride HS)]
__device__ __forceinline__ void layer_step(const half_t* __restrict__ WzT,
                                           const half_t* __restrict__ WhT, int K,
                                           const half_t* r0, int s0, int len0,
                                           half_t* lh /* state == region1 */,
                                           half_t* zbuf,
                                           float vz, float vh,
                                           int lane, int ncol, int rbase) {
  const int laneRow = lane & 15;
  const int laneC0  = (lane >> 4) << 3;   // A-chunk select
  const int laneK   = (lane >> 4) << 3;   // B-chunk select
  const half_t* pa0 = r0 + laneRow * s0 + laneC0;
  const half_t* pa1 = lh + laneRow * HS + laneC0;
  const int nk0 = len0 >> 5;
  const int nk1 = (K - len0) >> 5;

  v8f zero = {};
  v8f acc[4];

  // ---- z-gate pass ----
  {
    const half_t* wz = WzT + ncol * K + laneK;
    #pragma unroll
    for (int m = 0; m < 4; ++m) acc[m] = zero;
    gemm_region(wz,        pa0, s0, nk0, acc);
    gemm_region(wz + len0, pa1, HS, nk1, acc);
    #pragma unroll
    for (int m = 0; m < 4; ++m)
      #pragma unroll
      for (int r = 0; r < 8; ++r)
        zbuf[(m * 16 + rbase + r) * HS + ncol] = (half_t)sigmoidf_(acc[m][r] + vz);
  }

  // ---- candidate pass ----
  {
    const half_t* wh = WhT + ncol * K + laneK;
    #pragma unroll
    for (int m = 0; m < 4; ++m) acc[m] = zero;
    gemm_region(wh,        pa0, s0, nk0, acc);
    gemm_region(wh + len0, pa1, HS, nk1, acc);
  }

  __syncthreads();  // all cross-wave A reads of lh complete

  // owner-lane read-modify-write of h (unique (row,ncol) per lane -> race-free)
  #pragma unroll
  for (int m = 0; m < 4; ++m)
    #pragma unroll
    for (int r = 0; r < 8; ++r) {
      const int idx = (m * 16 + rbase + r) * HS + ncol;
      float z  = (float)zbuf[idx];
      float ho = (float)lh[idx];
      float c  = tanhf(acc[m][r] + vh);
      lh[idx] = (half_t)(z * ho + (1.0f - z) * c);
    }
  __syncthreads();  // h_new visible to everyone
}

// one-time: W[K][512] f32 -> WT[512][K] f16 (coalesced writes)
__global__ void cvt_transpose_kernel(const float* __restrict__ W, half_t* __restrict__ WT, int K) {
  int idx = blockIdx.x * 256 + threadIdx.x;
  if (idx >= K * HH) return;
  int n = idx / K;
  int k = idx - n * K;
  WT[idx] = (half_t)W[k * HH + n];
}

__global__ __launch_bounds__(1024)
void ugrnn_persistent_kernel(const float* __restrict__ x,
                             const float* __restrict__ bz0, const float* __restrict__ bh0,
                             const float* __restrict__ bz1, const float* __restrict__ bh1,
                             const half_t* __restrict__ WzT0, const half_t* __restrict__ WhT0,
                             const half_t* __restrict__ WzT1, const half_t* __restrict__ WhT1,
                             const float* __restrict__ Wfc, const float* __restrict__ bfc,
                             float* __restrict__ out) {
  __shared__ __align__(16) half_t lx[BB * XS];     //  33.8 KB  staged x_t
  __shared__ __align__(16) half_t lh0[BB * HS];    //  66.6 KB  layer-0 hidden state
  __shared__ __align__(16) half_t lh1[BB * HS];    //  66.6 KB  layer-1 hidden state
  __shared__ __align__(16) half_t zbuf[BB * HS];   //  66.6 KB  z-gate staging (per-lane private)

  const int tid   = threadIdx.x;
  const int w     = tid >> 5;               // wave id 0..31 -> owns N-tile w
  const int lane  = tid & 31;
  const int ncol  = w * 16 + (lane & 15);   // global output column for this lane
  const int rbase = (lane >> 4) << 3;       // C-layout: lanes>=16 hold M rows 8..15

  for (int i = tid; i < BB * HS; i += 1024) { lh0[i] = (half_t)0.0f; lh1[i] = (half_t)0.0f; }

  const float vz0 = bz0[ncol], vh0 = bh0[ncol];
  const float vz1 = bz1[ncol], vh1 = bh1[ncol];

  __syncthreads();

  for (int t = 0; t < SS; ++t) {
    // ---- stage x_t into LDS as f16 (coalesced: 16 floats / thread) ----
    {
      int row = tid >> 4;
      int c0  = (tid & 15) * 16;
      const float* xp = x + (row * SS + t) * FIN + c0;
      half_t* dst = lx + row * XS + c0;
      #pragma unroll
      for (int j = 0; j < 16; j += 4) {
        v4f v = *(const v4f*)(xp + j);
        dst[j + 0] = (half_t)v.x;
        dst[j + 1] = (half_t)v.y;
        dst[j + 2] = (half_t)v.z;
        dst[j + 3] = (half_t)v.w;
      }
    }
    __syncthreads();

    // layer 0: comb = [x_t | h0], K = 768
    layer_step(WzT0, WhT0, FIN + HH, lx, XS, FIN, lh0, zbuf, vz0, vh0, lane, ncol, rbase);
    // layer 1: comb = [h0_new | h1], K = 1024
    layer_step(WzT1, WhT1, HH + HH, lh0, HS, HH, lh1, zbuf, vz1, vh1, lane, ncol, rbase);
  }

  // ---- FC head on final h1 : out = h1 @ Wfc + bfc (64x128) ----
  {
    const int b  = tid >> 4;           // batch row 0..63
    const int c0 = (tid & 15) * 8;     // 8 output cols / thread
    float acc[8];
    #pragma unroll
    for (int j = 0; j < 8; ++j) acc[j] = bfc[c0 + j];
    for (int k = 0; k < HH; ++k) {
      float hv = (float)lh1[b * HS + k];
      v4f w0 = *(const v4f*)(Wfc + k * FOUT + c0);
      v4f w1 = *(const v4f*)(Wfc + k * FOUT + c0 + 4);
      acc[0] += hv * w0.x; acc[1] += hv * w0.y; acc[2] += hv * w0.z; acc[3] += hv * w0.w;
      acc[4] += hv * w1.x; acc[5] += hv * w1.y; acc[6] += hv * w1.z; acc[7] += hv * w1.w;
    }
    #pragma unroll
    for (int j = 0; j < 8; ++j) out[b * FOUT + c0 + j] = acc[j];
  }
}

extern "C" void kernel_launch(void* const* d_in, const int* in_sizes, int n_in,
                              void* d_out, int out_size, void* d_ws, size_t ws_size,
                              hipStream_t stream) {
  (void)in_sizes; (void)n_in; (void)out_size; (void)ws_size;

  const float* x   = (const float*)d_in[0];
  const float* Wz0 = (const float*)d_in[1];
  const float* bz0 = (const float*)d_in[2];
  const float* Wh0 = (const float*)d_in[3];
  const float* bh0 = (const float*)d_in[4];
  const float* Wz1 = (const float*)d_in[5];
  const float* bz1 = (const float*)d_in[6];
  const float* Wh1 = (const float*)d_in[7];
  const float* bh1 = (const float*)d_in[8];
  const float* Wfc = (const float*)d_in[9];
  const float* bfc = (const float*)d_in[10];
  float* out = (float*)d_out;

  // ws layout: f16 transposed weights [N=512][K]
  half_t* WzT0 = (half_t*)d_ws;                 // 512*768
  half_t* WhT0 = WzT0 + 512 * 768;              // 512*768
  half_t* WzT1 = WhT0 + 512 * 768;              // 512*1024
  half_t* WhT1 = WzT1 + 512 * 1024;             // 512*1024  (total ~3.5 MB)

  const int n0 = 768 * 512, n1 = 1024 * 512;
  cvt_transpose_kernel<<<(n0 + 255) / 256, 256, 0, stream>>>(Wz0, WzT0, 768);
  cvt_transpose_kernel<<<(n0 + 255) / 256, 256, 0, stream>>>(Wh0, WhT0, 768);
  cvt_transpose_kernel<<<(n1 + 255) / 256, 256, 0, stream>>>(Wz1, WzT1, 1024);
  cvt_transpose_kernel<<<(n1 + 255) / 256, 256, 0, stream>>>(Wh1, WhT1, 1024);

  ugrnn_persistent_kernel<<<1, 1024, 0, stream>>>(
      x, bz0, bh0, bz1, bh1, WzT0, WhT0, WzT1, WhT1, Wfc, bfc, out);
}
